// multiHeadedAttention_20495583936827
// MI455X (gfx1250) — matrix operand order repaired and checked by reading, not compiled, not run on hardware
//
#include <hip/hip_runtime.h>

typedef _Float16 f16;
typedef __attribute__((ext_vector_type(16))) _Float16 v16h;
typedef __attribute__((ext_vector_type(8)))  _Float16 v8h;
typedef __attribute__((ext_vector_type(8)))  float    v8f;

#define NB  2
#define SS  2048
#define EE  1024
#define HH  16
#define DKK 64

#if defined(__has_builtin)
#if __has_builtin(__builtin_amdgcn_tensor_load_to_lds) && \
    __has_builtin(__builtin_amdgcn_s_wait_tensorcnt)
#define USE_TDM 1
#endif
#endif
#ifndef USE_TDM
#define USE_TDM 0
#endif

static __device__ __forceinline__ v8f wmma_f16(v16h a, v16h b, v8f c) {
  return __builtin_amdgcn_wmma_f32_16x16x32_f16(false, a, false, b, (short)0, c, false, false);
}

static __device__ __forceinline__ v16h join_a(v8h lo, v8h hi) {
  v16h r;
#pragma unroll
  for (int e = 0; e < 8; ++e) { r[e] = lo[e]; r[e + 8] = hi[e]; }
  return r;
}

// A-fragment (16x32 f16): lane holds row m=lane&15; elements map to
// k = (lane>>4)*16 + e. p must point at row_base + k0 + (lane>>4)*16.
static __device__ __forceinline__ v16h load_a_f16(const f16* p) {
  v8h lo = *(const v8h*)(p);
  v8h hi = *(const v8h*)(p + 8);
  return join_a(lo, hi);
}

static __device__ __forceinline__ v16h load_a_f32cvt(const float* p) {
  float4 f0 = *(const float4*)(p);
  float4 f1 = *(const float4*)(p + 4);
  float4 f2 = *(const float4*)(p + 8);
  float4 f3 = *(const float4*)(p + 12);
  v16h a;
  a[0]  = (f16)f0.x; a[1]  = (f16)f0.y; a[2]  = (f16)f0.z; a[3]  = (f16)f0.w;
  a[4]  = (f16)f1.x; a[5]  = (f16)f1.y; a[6]  = (f16)f1.z; a[7]  = (f16)f1.w;
  a[8]  = (f16)f2.x; a[9]  = (f16)f2.y; a[10] = (f16)f2.z; a[11] = (f16)f2.w;
  a[12] = (f16)f3.x; a[13] = (f16)f3.y; a[14] = (f16)f3.z; a[15] = (f16)f3.w;
  return a;
}

#if USE_TDM
typedef __attribute__((ext_vector_type(4))) unsigned int u32x4;
typedef __attribute__((ext_vector_type(8))) int          i32x8;
typedef __attribute__((ext_vector_type(4))) int          i32x4;
typedef __attribute__((address_space(3))) const void*    lds_cvp;

static __device__ __forceinline__ unsigned lds_addr_of(const void* p) {
  return (unsigned)(unsigned long long)(lds_cvp)p;   // generic -> LDS offset
}

// 2D TDM load: tile (tile_w x tile_h) of 2-byte elements from a row-major
// tensor whose rows are stride_elems apart, into packed LDS at ldsOff.
// D# layout per CDNA5 ISA 8.3/8.4: group0 = {count/flags, lds_addr, global_addr, type=2},
// group1 = {data_size, tensor_dim0/1, tile_dim0/1, tensor_dim0_stride}.
static __device__ __forceinline__ void tdm_load_2d(
    unsigned ldsOff, const void* gaddr,
    unsigned tensor_w, unsigned tensor_h, unsigned stride_elems,
    unsigned tile_w, unsigned tile_h)
{
  unsigned long long ga = (unsigned long long)gaddr;
  u32x4 g0;
  g0[0] = 1u;                                   // count=1, is_restore=0, gather off
  g0[1] = ldsOff;                               // lds_addr [63:32]
  g0[2] = (unsigned)(ga & 0xFFFFFFFFu);         // global_addr lo
  g0[3] = (unsigned)(ga >> 32) | 0x80000000u;   // global_addr hi | type=2 << 30
  i32x8 g1;
  g1[0] = (int)(1u << 16);                      // data_size=1 (2 bytes), no multicast
  g1[1] = (int)((tensor_w & 0xFFFFu) << 16);                        // tensor_dim0 lo
  g1[2] = (int)((tensor_w >> 16) | ((tensor_h & 0xFFFFu) << 16));   // dim0 hi | dim1 lo
  g1[3] = (int)((tensor_h >> 16) | (tile_w << 16));                 // dim1 hi | tile_dim0
  g1[4] = (int)(tile_h & 0xFFFFu);                                  // tile_dim1 (tile_dim2=0)
  g1[5] = (int)stride_elems;                                        // tensor_dim0_stride lo
  g1[6] = 0;
  g1[7] = 0;
  i32x4 z4 = {0, 0, 0, 0};
  i32x8 z8 = {0, 0, 0, 0, 0, 0, 0, 0};
  // 6-arg form on this toolchain: (g0, g1, g2, g3, extra, cpol)
  __builtin_amdgcn_tensor_load_to_lds(g0, g1, z4, z4, z8, 0);
}
#endif

// ---------------- weight f32 -> f16 convert ----------------
__global__ void mha_cvt_w(const float* __restrict__ w, f16* __restrict__ dst) {
  int i = blockIdx.x * blockDim.x + threadIdx.x;
  dst[i] = (f16)w[i];
}

// ---------------- QKV projection: Y = X @ W^T + b, scattered to head layout ----------------
__global__ void __launch_bounds__(128) mha_qkv_proj(
    const float* __restrict__ Xq, const float* __restrict__ Xk, const float* __restrict__ Xv,
    const f16* __restrict__ W1h, const f16* __restrict__ W2h, const f16* __restrict__ W3h,
    const float* __restrict__ b1, const float* __restrict__ b2, const float* __restrict__ b3,
    f16* __restrict__ qh, f16* __restrict__ kh, f16* __restrict__ vt)
{
  const int mode = blockIdx.y;                 // 0=Q, 1=K, 2=V
  const float* X    = (mode == 0) ? Xq : (mode == 1) ? Xk : Xv;
  const f16*   W    = (mode == 0) ? W1h : (mode == 1) ? W2h : W3h;
  const float* bias = (mode == 0) ? b1 : (mode == 1) ? b2 : b3;

  const int wid  = threadIdx.x >> 5;
  const int lane = threadIdx.x & 31;
  const int nl = lane & 15, hf = lane >> 4;
  const int NT = EE / 64;                      // 16 column tiles
  int wt  = blockIdx.x * 4 + wid;              // 0..2047
  int mt  = wt / NT, nt = wt % NT;
  int row0 = mt * 32, col0 = nt * 64;

  v8f acc[2][4];
#pragma unroll
  for (int i = 0; i < 2; ++i)
#pragma unroll
    for (int j = 0; j < 4; ++j) acc[i][j] = {};

  for (int k0 = 0; k0 < EE; k0 += 32) {
    v16h a[2];
#pragma unroll
    for (int i = 0; i < 2; ++i)
      a[i] = load_a_f32cvt(X + (size_t)(row0 + 16 * i + nl) * EE + k0 + hf * 16);
    v16h bm[4];
#pragma unroll
    for (int j = 0; j < 4; ++j)   // B(k,n) = W[n*E + k], contiguous 32B per lane
      bm[j] = *(const v16h*)(W + (size_t)(col0 + 16 * j + nl) * EE + k0 + hf * 16);
#pragma unroll
    for (int i = 0; i < 2; ++i)
#pragma unroll
      for (int j = 0; j < 4; ++j)
        acc[i][j] = wmma_f16(a[i], bm[j], acc[i][j]);
  }

#pragma unroll
  for (int i = 0; i < 2; ++i) {
#pragma unroll
    for (int j = 0; j < 4; ++j) {
      int col = col0 + 16 * j + nl;
      int h = col >> 6, d = col & 63;
      float bv = bias[col];
      if (mode == 2) {
        // V transposed store: 8 consecutive s positions -> one 16B store
        v8h pk;
#pragma unroll
        for (int r = 0; r < 8; ++r) pk[r] = (f16)(acc[i][j][r] + bv);
        int rowb = row0 + 16 * i + hf * 8;
        int bb = rowb >> 11, s = rowb & (SS - 1);
        *(v8h*)(vt + ((size_t)(bb * HH + h) * DKK + d) * SS + s) = pk;
      } else {
#pragma unroll
        for (int r = 0; r < 8; ++r) {
          int row = row0 + 16 * i + r + hf * 8;
          int bb = row >> 11, s = row & (SS - 1);
          float val = acc[i][j][r] + bv;
          if (mode == 0)      // fold 1/sqrt(DK)=0.125 into stored Q
            qh[((size_t)(bb * HH + h) * SS + s) * DKK + d] = (f16)(val * 0.125f);
          else
            kh[((size_t)(bb * HH + h) * SS + s) * DKK + d] = (f16)val;
        }
      }
    }
  }
}

// ---------------- flash attention ----------------
// Block = 4 waves = 64 query rows of one (b,h). K/V chunks staged in LDS by the
// Tensor Data Mover (wave 0 issues, TENSORcnt-waited, block barrier), shared by
// all 4 waves. P is re-laid-out C->A through per-wave LDS (same-wave LDS ops are
// in-order on CDNA5).
__global__ void __launch_bounds__(128) mha_flash(
    const f16* __restrict__ qh, const f16* __restrict__ kh, const f16* __restrict__ vt,
    const int* __restrict__ mask, f16* __restrict__ attnh)
{
  __shared__ __align__(32) f16 Kt[32 * DKK];      // 32 keys x 64 dk
  __shared__ __align__(32) f16 Vt[DKK * 32];      // 64 dk rows x 32 keys
  __shared__ __align__(32) f16 pbuf[4][16 * 32];  // per-wave P staging
  const int wid  = threadIdx.x >> 5;
  const int lane = threadIdx.x & 31;
  const int nl = lane & 15, hf = lane >> 4;

  int blk = blockIdx.x;                   // NB*HH*(SS/64) blocks
  int qb = blk & (SS / 64 - 1);           // 32 query blocks of 64 rows
  int bh = blk >> 5;
  int h = bh & (HH - 1), bb = bh >> 4;
  int q0 = qb * 64 + wid * 16;

  const f16* Qb  = qh + (size_t)(bb * HH + h) * SS * DKK;
  const f16* Kb  = kh + (size_t)(bb * HH + h) * SS * DKK;
  const f16* Vbg = vt + (size_t)(bb * HH + h) * DKK * SS;
  const int* Mb  = mask + bb * SS;

  v16h qa0 = load_a_f16(Qb + (size_t)(q0 + nl) * DKK + hf * 16);
  v16h qa1 = load_a_f16(Qb + (size_t)(q0 + nl) * DKK + 32 + hf * 16);

  float m_i[8], l_i[8];
  v8f o[4];
#pragma unroll
  for (int r = 0; r < 8; ++r) { m_i[r] = -3.0e38f; l_i[r] = 0.f; }
#pragma unroll
  for (int j = 0; j < 4; ++j) o[j] = {};

  f16* pl = pbuf[wid];
#if USE_TDM
  const unsigned ktOff = lds_addr_of(Kt);
  const unsigned vtOff = lds_addr_of(Vt);
#endif

  for (int kc = 0; kc < SS; kc += 32) {
#if USE_TDM
    if (wid == 0) {
      // K chunk: 32 rows x 64 elems, row stride 64
      tdm_load_2d(ktOff, Kb + (size_t)kc * DKK, DKK, 32, DKK, DKK, 32);
      // V chunk: 64 rows x 32 elems, row stride S
      tdm_load_2d(vtOff, Vbg + kc, 32, DKK, SS, 32, DKK);
      __builtin_amdgcn_s_wait_tensorcnt(0);
    }
#else
    {
      const unsigned* ks = (const unsigned*)(Kb + (size_t)kc * DKK);
      unsigned* kd = (unsigned*)Kt;
      for (int t = threadIdx.x; t < 32 * DKK / 2; t += 128) kd[t] = ks[t];
      unsigned* vd = (unsigned*)Vt;
      for (int t = threadIdx.x; t < DKK * 16; t += 128) {
        int d = t >> 4, cw = t & 15;
        vd[d * 16 + cw] = ((const unsigned*)(Vbg + (size_t)d * SS + kc))[cw];
      }
    }
#endif
    __syncthreads();

    v8f s0 = {}, s1 = {};
    {
      v16h b00 = *(const v16h*)(Kt + nl * DKK + hf * 16);
      v16h b01 = *(const v16h*)(Kt + nl * DKK + 32 + hf * 16);
      s0 = wmma_f16(qa0, b00, s0);
      s0 = wmma_f16(qa1, b01, s0);
      v16h b10 = *(const v16h*)(Kt + (16 + nl) * DKK + hf * 16);
      v16h b11 = *(const v16h*)(Kt + (16 + nl) * DKK + 32 + hf * 16);
      s1 = wmma_f16(qa0, b10, s1);
      s1 = wmma_f16(qa1, b11, s1);
    }
    float mk0 = (Mb[kc + nl] == 0)      ? -1.0e9f : 0.f;
    float mk1 = (Mb[kc + 16 + nl] == 0) ? -1.0e9f : 0.f;

#pragma unroll
    for (int r = 0; r < 8; ++r) {
      float sv0 = s0[r] + mk0;
      float sv1 = s1[r] + mk1;
      float cm = fmaxf(sv0, sv1);
#pragma unroll
      for (int msk = 1; msk < 16; msk <<= 1)       // row reduce within 16-lane halves
        cm = fmaxf(cm, __shfl_xor(cm, msk, 32));
      float mnew = fmaxf(m_i[r], cm);
      float sc = __expf(m_i[r] - mnew);
      float p0 = __expf(sv0 - mnew);
      float p1 = __expf(sv1 - mnew);
      float rs = p0 + p1;
#pragma unroll
      for (int msk = 1; msk < 16; msk <<= 1)
        rs += __shfl_xor(rs, msk, 32);
      l_i[r] = l_i[r] * sc + rs;
      m_i[r] = mnew;
#pragma unroll
      for (int j = 0; j < 4; ++j) o[j][r] *= sc;
      pl[(r + hf * 8) * 32 + nl]      = (f16)p0;   // C-layout -> row-major stage
      pl[(r + hf * 8) * 32 + 16 + nl] = (f16)p1;
    }

    v16h pa;
    {
      const f16* pp = pl + nl * 32 + hf * 16;
      v8h lo = *(const v8h*)pp;
      v8h hi = *(const v8h*)(pp + 8);
      pa = join_a(lo, hi);
    }
#pragma unroll
    for (int j = 0; j < 4; ++j) {
      v16h bv = *(const v16h*)(Vt + (16 * j + nl) * 32 + hf * 16);
      o[j] = wmma_f16(pa, bv, o[j]);
    }
    __syncthreads();   // protect Kt/Vt before next chunk overwrite
  }

#pragma unroll
  for (int r = 0; r < 8; ++r) {
    float inv = 1.f / l_i[r];
    int row = q0 + r + hf * 8;
#pragma unroll
    for (int j = 0; j < 4; ++j) {
      int col = h * DKK + 16 * j + nl;
      attnh[((size_t)bb * SS + row) * EE + col] = (f16)(o[j][r] * inv);
    }
  }
}

// ---------------- output projection: out = A @ W4^T + b4 (f32 out) ----------------
__global__ void __launch_bounds__(128) mha_out_proj(
    const f16* __restrict__ Ah, const f16* __restrict__ W4h,
    const float* __restrict__ b4, float* __restrict__ out)
{
  const int wid  = threadIdx.x >> 5;
  const int lane = threadIdx.x & 31;
  const int nl = lane & 15, hf = lane >> 4;
  const int NT = EE / 64;
  int wt  = blockIdx.x * 4 + wid;
  int mt  = wt / NT, nt = wt % NT;
  int row0 = mt * 32, col0 = nt * 64;

  v8f acc[2][4];
#pragma unroll
  for (int i = 0; i < 2; ++i)
#pragma unroll
    for (int j = 0; j < 4; ++j) acc[i][j] = {};

  for (int k0 = 0; k0 < EE; k0 += 32) {
    v16h a[2];
#pragma unroll
    for (int i = 0; i < 2; ++i)
      a[i] = load_a_f16(Ah + (size_t)(row0 + 16 * i + nl) * EE + k0 + hf * 16);
    v16h bm[4];
#pragma unroll
    for (int j = 0; j < 4; ++j)
      bm[j] = *(const v16h*)(W4h + (size_t)(col0 + 16 * j + nl) * EE + k0 + hf * 16);
#pragma unroll
    for (int i = 0; i < 2; ++i)
#pragma unroll
      for (int j = 0; j < 4; ++j)
        acc[i][j] = wmma_f16(a[i], bm[j], acc[i][j]);
  }

#pragma unroll
  for (int i = 0; i < 2; ++i) {
#pragma unroll
    for (int j = 0; j < 4; ++j) {
      int col = col0 + 16 * j + nl;
      float bv = b4[col];
#pragma unroll
      for (int r = 0; r < 8; ++r) {
        int row = row0 + 16 * i + r + hf * 8;
        out[(size_t)row * EE + col] = acc[i][j][r] + bv;
      }
    }
  }
}

extern "C" void kernel_launch(void* const* d_in, const int* in_sizes, int n_in,
                              void* d_out, int out_size, void* d_ws, size_t ws_size,
                              hipStream_t stream) {
  (void)in_sizes; (void)n_in; (void)out_size; (void)ws_size;
  const float* q   = (const float*)d_in[0];
  const float* k   = (const float*)d_in[1];
  const float* v   = (const float*)d_in[2];
  const int*   msk = (const int*)d_in[3];
  const float* W1  = (const float*)d_in[4];
  const float* b1  = (const float*)d_in[5];
  const float* W2  = (const float*)d_in[6];
  const float* b2  = (const float*)d_in[7];
  const float* W3  = (const float*)d_in[8];
  const float* b3  = (const float*)d_in[9];
  const float* W4  = (const float*)d_in[10];
  const float* b4  = (const float*)d_in[11];

  // workspace carve-up: 4x 8MiB f16 activations + 4x 2MiB f16 weights = 40MiB
  const size_t actN = (size_t)NB * HH * SS * DKK;   // 4Mi elements
  f16* qh    = (f16*)d_ws;
  f16* kh    = qh + actN;
  f16* vt    = kh + actN;
  f16* attnh = vt + actN;
  f16* W1h   = attnh + actN;
  f16* W2h   = W1h + (size_t)EE * EE;
  f16* W3h   = W2h + (size_t)EE * EE;
  f16* W4h   = W3h + (size_t)EE * EE;

  mha_cvt_w<<<EE * EE / 256, 256, 0, stream>>>(W1, W1h);
  mha_cvt_w<<<EE * EE / 256, 256, 0, stream>>>(W2, W2h);
  mha_cvt_w<<<EE * EE / 256, 256, 0, stream>>>(W3, W3h);
  mha_cvt_w<<<EE * EE / 256, 256, 0, stream>>>(W4, W4h);

  // 128 row-tiles * 16 col-tiles = 2048 wave tiles, 4 waves/block
  mha_qkv_proj<<<dim3(512, 3), 128, 0, stream>>>(q, k, v, W1h, W2h, W3h,
                                                 b1, b2, b3, qh, kh, vt);
  // one block per (b,h,64 query rows)
  mha_flash<<<NB * HH * (SS / 64), 128, 0, stream>>>(qh, kh, vt, msk, attnh);
  mha_out_proj<<<512, 128, 0, stream>>>(attnh, W4h, b4, (float*)d_out);
}
